// QKV_6227702579510
// MI455X (gfx1250) — matrix operand order repaired and checked by reading, not compiled, not run on hardware
//
#include <hip/hip_runtime.h>
#include <hip/hip_bf16.h>

// ---------------------------------------------------------------------------
// Graph attention (segment softmax) for MI455X / gfx1250.
// GEMMs run on v_wmma_f32_16x16x32_bf16; edge tiles staged to LDS with the
// gfx1250 async-to-LDS engine (ASYNCcnt) when the builtin is available.
// ---------------------------------------------------------------------------

typedef __attribute__((ext_vector_type(16))) __bf16 v16bf;
typedef __attribute__((ext_vector_type(8)))  __bf16 v8bf;
typedef __attribute__((ext_vector_type(8)))  float  v8f;

#define GAT_HID   128
#define GAT_HEADS 4

#if __has_builtin(__builtin_amdgcn_global_load_async_to_lds_b128)
#define GAT_ASYNC 1
// Builtin parameter types (from hipcc diagnostic): pointee is a gcc-style
// vector_size(16) int; global operand in AS1, LDS operand in AS3.
typedef int gat_v4i __attribute__((vector_size(16)));
typedef __attribute__((address_space(1))) gat_v4i* gat_gptr;
typedef __attribute__((address_space(3))) gat_v4i* gat_lptr;
#endif

__device__ __forceinline__ void gat_wait_async() {
#if defined(GAT_ASYNC)
#if __has_builtin(__builtin_amdgcn_s_wait_asynccnt)
    __builtin_amdgcn_s_wait_asynccnt(0);
#else
    asm volatile("s_wait_asynccnt 0" ::: "memory");
#endif
#endif
}

// ---- WMMA fragment helpers -------------------------------------------------
// A fragment (16x32 bf16, lane = row M, halves split K):
//   lane<16 : a[0..7]=A[M][k0..k0+7],   a[8..15]=A[M][k0+16..k0+23]
//   lane>=16: a[0..7]=A[M][k0+8..k0+15],a[8..15]=A[M][k0+24..k0+31]
__device__ __forceinline__ v16bf load_a_frag_bf16(const __bf16* ldsX, int l16, int half, int k0) {
    const __bf16* p0 = ldsX + l16 * GAT_HID + k0 + half * 8;
    v8bf lo = *(const v8bf*)p0;
    v8bf hi = *(const v8bf*)(p0 + 16);
    return __builtin_shufflevector(lo, hi, 0,1,2,3,4,5,6,7,8,9,10,11,12,13,14,15);
}

__device__ __forceinline__ v8bf cvt8_bf16(v8f x) {
    v8bf r;
#pragma unroll
    for (int i = 0; i < 8; ++i) r[i] = (__bf16)x[i];
    return r;
}

// Same A fragment but sourced from an f32 tile in LDS (async-staged path).
__device__ __forceinline__ v16bf load_a_frag_f32(const float* ldsX, int l16, int half, int k0) {
    const float* p0 = ldsX + l16 * GAT_HID + k0 + half * 8;
    v8f lo = *(const v8f*)p0;          // 2x ds_load_b128
    v8f hi = *(const v8f*)(p0 + 16);   // 2x ds_load_b128
    return __builtin_shufflevector(cvt8_bf16(lo), cvt8_bf16(hi),
                                   0,1,2,3,4,5,6,7,8,9,10,11,12,13,14,15);
}

// B fragment (32x16 bf16, lane = col N, same K split as A fragment:
// b[0..7]=k0+8*half.., b[8..15]=k0+16+8*half..).
// Y = X @ W^T, so B[k][n] = W[colbase+n][k]: two aligned 16-byte loads.
__device__ __forceinline__ v16bf load_b_frag(const __bf16* Wg, int col, int half, int k0) {
    const __bf16* p0 = Wg + (size_t)col * GAT_HID + k0 + half * 8;
    v8bf lo = *(const v8bf*)p0;
    v8bf hi = *(const v8bf*)(p0 + 16);
    return __builtin_shufflevector(lo, hi, 0,1,2,3,4,5,6,7,8,9,10,11,12,13,14,15);
}

// One wave: 16x16 f32 tile of (X_tile[16x128] @ W^T[:, colbase:colbase+16]),
// A tile in LDS as bf16.
__device__ __forceinline__ v8f wmma_tile_bf16(const __bf16* ldsX, const __bf16* Wg,
                                              int colbase, int lane) {
    const int half = lane >> 4, l16 = lane & 15;
    v8f c = {};
#pragma unroll
    for (int k0 = 0; k0 < GAT_HID; k0 += 32) {
        v16bf a = load_a_frag_bf16(ldsX, l16, half, k0);
        v16bf b = load_b_frag(Wg, colbase + l16, half, k0);
        c = __builtin_amdgcn_wmma_f32_16x16x32_bf16(false, a, false, b,
                                                    (short)0, c, false, false);
    }
    return c;
}

// Same but A tile in LDS as f32 (converted in-register).
__device__ __forceinline__ v8f wmma_tile_f32(const float* ldsX, const __bf16* Wg,
                                             int colbase, int lane) {
    const int half = lane >> 4, l16 = lane & 15;
    v8f c = {};
#pragma unroll
    for (int k0 = 0; k0 < GAT_HID; k0 += 32) {
        v16bf a = load_a_frag_f32(ldsX, l16, half, k0);
        v16bf b = load_b_frag(Wg, colbase + l16, half, k0);
        c = __builtin_amdgcn_wmma_f32_16x16x32_bf16(false, a, false, b,
                                                    (short)0, c, false, false);
    }
    return c;
}

// Stage a 16x128 f32 tile into LDS: async-to-LDS engine when available.
__device__ __forceinline__ void stage_tile_f32(const float* __restrict__ gsrc,
                                               float* ldsDst, int tid) {
#if defined(GAT_ASYNC)
#pragma unroll
    for (int c = 0; c < 2; ++c) {               // 512 b128 chunks / 256 threads
        int chunk = tid + c * 256;
        __builtin_amdgcn_global_load_async_to_lds_b128(
            (gat_gptr)(gsrc + chunk * 4),
            (gat_lptr)(ldsDst + chunk * 4), 0, 0);
    }
    gat_wait_async();
#else
    for (int i = tid; i < 16 * GAT_HID; i += 256) ldsDst[i] = gsrc[i];
#endif
}

// Float atomic max via signed/unsigned integer reinterpretation (init = -inf).
__device__ __forceinline__ void atomicMaxF(float* addr, float v) {
    if (v >= 0.0f) atomicMax((int*)addr, __float_as_int(v));
    else           atomicMin((unsigned int*)addr, (unsigned int)__float_as_int(v));
}

// ---- Kernel 0: convert the 4 weight matrices to bf16 (128KB, lives in L2) ---
__global__ void gat_convert_weights(const float* __restrict__ Wq,
                                    const float* __restrict__ Wk,
                                    const float* __restrict__ Wv,
                                    const float* __restrict__ Wo,
                                    __bf16* __restrict__ out) {
    int i = blockIdx.x * 256 + threadIdx.x;      // 4 * 16384 = 65536 total
    const float* srcs[4] = {Wq, Wk, Wv, Wo};
    int m = i >> 14, r = i & 16383;
    out[i] = (__bf16)srcs[m][r];
}

// ---- Kernel 1: init m=-inf, s=0, agg=0 -------------------------------------
__global__ void gat_init(float* __restrict__ m, float* __restrict__ s,
                         float* __restrict__ agg, int N) {
    int i = blockIdx.x * 256 + threadIdx.x;
    if (i < N * GAT_HEADS) { m[i] = -__builtin_inff(); s[i] = 0.0f; }
    if (i < N * GAT_HID)   agg[i] = 0.0f;
}

// ---- Kernel 2/6: node GEMM  Y[N,128] = X[N,128] @ W^T ----------------------
__global__ void gat_node_gemm(const float* __restrict__ X, const __bf16* __restrict__ W,
                              float* __restrict__ Y) {
    __shared__ __attribute__((aligned(32))) __bf16 ldsX[16 * GAT_HID];
    const int rowbase = blockIdx.x * 16;
    const int tid = threadIdx.x;
    for (int i = tid; i < 16 * GAT_HID; i += 256) {
        int r = i >> 7, k = i & 127;
        ldsX[i] = (__bf16)X[(size_t)(rowbase + r) * GAT_HID + k];
    }
    __syncthreads();
    const int wave = tid >> 5, lane = tid & 31;
    v8f c = wmma_tile_bf16(ldsX, W, wave * 16, lane);
    const int half = lane >> 4, l16 = lane & 15;
#pragma unroll
    for (int i = 0; i < 8; ++i)   // C layout: lane=col, rows = 8*half + i
        Y[(size_t)(rowbase + half * 8 + i) * GAT_HID + wave * 16 + l16] = c[i];
}

// ---- Kernel 3: edge pass 1 — K tile (WMMA) -> logits -> atomic max ---------
__global__ void gat_edge_logits(const float* __restrict__ hE, const __bf16* __restrict__ Wk,
                                const float* __restrict__ Qn, const int* __restrict__ center,
                                float* __restrict__ logits, float* __restrict__ mbuf) {
    __shared__ __attribute__((aligned(32))) float ldsE[16 * GAT_HID];  // f32 tile
    __shared__ float ldsK[16][GAT_HID];
    const int e0 = blockIdx.x * 16;
    const int tid = threadIdx.x;
    __builtin_prefetch(hE + (size_t)(e0 + 16) * GAT_HID, 0, 1);  // global_prefetch_b8
    stage_tile_f32(hE + (size_t)e0 * GAT_HID, ldsE, tid);
    __syncthreads();
    const int wave = tid >> 5, lane = tid & 31;
    v8f c = wmma_tile_f32(ldsE, Wk, wave * 16, lane);
    const int half = lane >> 4, l16 = lane & 15;
#pragma unroll
    for (int i = 0; i < 8; ++i)
        ldsK[half * 8 + i][wave * 16 + l16] = c[i];
    __syncthreads();
    if (tid < 64) {                       // 16 edges x 4 heads
        int e = tid >> 2, h = tid & 3;
        int edge = e0 + e;
        int n = center[edge];
        const float* q = Qn + (size_t)n * GAT_HID + h * 32;
        float acc = 0.0f;
#pragma unroll
        for (int d = 0; d < 32; ++d) acc += q[d] * ldsK[e][h * 32 + d];
        acc *= 0.17677669529663687f;      // 1/sqrt(32)
        logits[(size_t)edge * GAT_HEADS + h] = acc;
        atomicMaxF(&mbuf[n * GAT_HEADS + h], acc);
    }
}

// ---- Kernel 4: e = exp(logit - m[center]); s[center] += e ------------------
__global__ void gat_softmax_num(float* __restrict__ logits, const float* __restrict__ mbuf,
                                float* __restrict__ sbuf, const int* __restrict__ center,
                                int E4) {
    int i = blockIdx.x * 256 + threadIdx.x;
    if (i >= E4) return;
    int edge = i >> 2, h = i & 3;
    int n = center[edge];
    float e = __expf(logits[i] - mbuf[n * GAT_HEADS + h]);
    logits[i] = e;
    atomicAdd(&sbuf[n * GAT_HEADS + h], e);
}

// ---- Kernel 5: edge pass 2 — V tile (WMMA), scale by attend, scatter-add ---
__global__ void gat_edge_agg(const float* __restrict__ hE, const __bf16* __restrict__ Wv,
                             const float* __restrict__ elog, const float* __restrict__ sbuf,
                             const int* __restrict__ center, float* __restrict__ agg) {
    __shared__ __attribute__((aligned(32))) float ldsE[16 * GAT_HID];  // f32 tile
    __shared__ float att[16][GAT_HEADS];
    __shared__ int   ctr[16];
    const int e0 = blockIdx.x * 16;
    const int tid = threadIdx.x;
    __builtin_prefetch(hE + (size_t)(e0 + 16) * GAT_HID, 0, 1);
    stage_tile_f32(hE + (size_t)e0 * GAT_HID, ldsE, tid);
    if (tid < 64) {
        int e = tid >> 2, h = tid & 3;
        int n = center[e0 + e];
        if (h == 0) ctr[e] = n;
        att[e][h] = elog[(size_t)(e0 + e) * GAT_HEADS + h] / sbuf[n * GAT_HEADS + h];
    }
    __syncthreads();
    const int wave = tid >> 5, lane = tid & 31;
    v8f c = wmma_tile_f32(ldsE, Wv, wave * 16, lane);
    const int half = lane >> 4, l16 = lane & 15;
    const int col = wave * 16 + l16;
    const int h = col >> 5;               // head of this output column
#pragma unroll
    for (int i = 0; i < 8; ++i) {
        int e = half * 8 + i;             // edge row within the tile
        float v = c[i] * att[e][h];
        atomicAdd(&agg[(size_t)ctr[e] * GAT_HID + col], v);
    }
}

// ---------------------------------------------------------------------------
extern "C" void kernel_launch(void* const* d_in, const int* in_sizes, int n_in,
                              void* d_out, int out_size, void* d_ws, size_t ws_size,
                              hipStream_t stream) {
    const float* h_V    = (const float*)d_in[0];
    const float* h_E    = (const float*)d_in[1];
    const int*   center = (const int*)  d_in[2];
    // d_in[3] batch_id, d_in[4] dst_idx: unused by the reference
    const float* W_Q    = (const float*)d_in[5];
    const float* W_K    = (const float*)d_in[6];
    const float* W_V    = (const float*)d_in[7];
    const float* W_O    = (const float*)d_in[8];
    float* out = (float*)d_out;

    const int N = in_sizes[0] / GAT_HID;     // 50000
    const int E = in_sizes[1] / GAT_HID;     // 1600000

    // ---- workspace layout (all 256B aligned) ----
    char* ws = (char*)d_ws;
    __bf16* Wbf   = (__bf16*)ws;                       // 4 * 16384 bf16 = 128 KiB
    size_t off    = 4 * 16384 * sizeof(__bf16);
    float* Qn     = (float*)(ws + off); off += (size_t)N * GAT_HID   * sizeof(float);
    float* logits = (float*)(ws + off); off += (size_t)E * GAT_HEADS * sizeof(float);
    float* mbuf   = (float*)(ws + off); off += (size_t)N * GAT_HEADS * sizeof(float);
    float* sbuf   = (float*)(ws + off); off += (size_t)N * GAT_HEADS * sizeof(float);
    float* agg    = (float*)(ws + off);
    __bf16* Wq_bf = Wbf + 0 * 16384;
    __bf16* Wk_bf = Wbf + 1 * 16384;
    __bf16* Wv_bf = Wbf + 2 * 16384;
    __bf16* Wo_bf = Wbf + 3 * 16384;

    // 0) weights -> bf16
    gat_convert_weights<<<256, 256, 0, stream>>>(W_Q, W_K, W_V, W_O, Wbf);
    // 1) init m/s/agg
    gat_init<<<(N * GAT_HID + 255) / 256, 256, 0, stream>>>(mbuf, sbuf, agg, N);
    // 2) Qn = h_V @ W_Q^T          (N = 50000 = 3125 * 16)
    gat_node_gemm<<<N / 16, 256, 0, stream>>>(h_V, Wq_bf, Qn);
    // 3) K tiles + logits + segment max   (E = 1.6M = 100000 * 16)
    gat_edge_logits<<<E / 16, 256, 0, stream>>>(h_E, Wk_bf, Qn, center, logits, mbuf);
    // 4) exp + segment sum
    gat_softmax_num<<<(E * GAT_HEADS + 255) / 256, 256, 0, stream>>>(logits, mbuf, sbuf,
                                                                     center, E * GAT_HEADS);
    // 5) V tiles, scale by attention, scatter-add into agg
    gat_edge_agg<<<E / 16, 256, 0, stream>>>(h_E, Wv_bf, logits, sbuf, center, agg);
    // 6) out = agg @ W_O^T
    gat_node_gemm<<<N / 16, 256, 0, stream>>>(agg, Wo_bf, out);
}
